// EpisodicMemory_18640158065230
// MI455X (gfx1250) — matrix operand order repaired and checked by reading, not compiled, not run on hardware
//
#include <hip/hip_runtime.h>
#include <math.h>

#define HID     1024
#define TWO_HID 2048
#define KD      32
#define SLOTS   64
#define MTILE   32          // rows of the batch per block (2 WMMA M-tiles)
#define THREADS 512         // 16 wave32
#define COMBS   2056        // comb row stride (ushorts): 1028 dwords = 4 mod 64 banks
#define ATTNS   72          // attn row stride (ushorts): 36 dwords, conflict-free

typedef __attribute__((ext_vector_type(16))) __bf16 v16bf;
typedef __attribute__((ext_vector_type(8)))  float  v8f;

union BFrag {
    v16bf v;
    unsigned short u[16];
    uint4 q[2];
};

__device__ __forceinline__ unsigned short f2bf(float f) {
    unsigned u = __float_as_uint(f);
    u += 0x7FFFu + ((u >> 16) & 1u);   // round-to-nearest-even
    return (unsigned short)(u >> 16);
}

__device__ __forceinline__ unsigned pack2bf(float lo, float hi) {
    return (unsigned)f2bf(lo) | ((unsigned)f2bf(hi) << 16);
}

__device__ __forceinline__ float gelu_exact(float x) {
    return 0.5f * x * (1.0f + erff(x * 0.70710678118654752440f));
}

// ---------------------------------------------------------------------------
// Prep 1: normalized keys (mem_keys + 0.1*pos_table[slot_order]) and salience
// bias vector. One block of 64 threads, one slot each.
// ---------------------------------------------------------------------------
__global__ void prep_keys_bias(const float* __restrict__ pos_table,
                               const float* __restrict__ mem_keys,
                               const float* __restrict__ mem_age,
                               const float* __restrict__ mem_conf,
                               const int*   __restrict__ slot_order,
                               float* __restrict__ keys_pos,
                               float* __restrict__ biasv) {
    __shared__ float mf[SLOTS];
    int s = threadIdx.x;
    float age  = mem_age[s];
    float freq = fmaxf(age, 1.0f);
    mf[s] = freq;
    __syncthreads();
    if (s == 0) {
        float m = mf[0];
        for (int i = 1; i < SLOTS; ++i) m = fmaxf(m, mf[i]);
        mf[0] = m;
    }
    __syncthreads();
    float fmax_ = mf[0];

    int so = slot_order[s] & (SLOTS - 1);
    float v[KD];
    float ss = 0.f;
    for (int d = 0; d < KD; ++d) {
        float t = mem_keys[s * KD + d] + 0.1f * pos_table[so * KD + d];
        v[d] = t;
        ss += t * t;
    }
    float inv = 1.0f / fmaxf(sqrtf(ss), 1e-12f);
    for (int d = 0; d < KD; ++d) keys_pos[s * KD + d] = v[d] * inv;

    float rec = expf(-age / 200.0f);
    float fn  = logf(freq + 1.0f) / (logf(fmax_ + 2.0f) + 1e-8f);
    biasv[s] = 0.2f * rec + 0.15f * fn + 0.1f * mem_conf[s] + 0.08f;  // 0.1*0.8
}

// ---------------------------------------------------------------------------
// Prep 2: repack a B operand into bf16 WMMA-fragment-major layout.
// Fragment f = ntile*(K/32) + ktile. Lane l holds B[kbase..kbase+15][n] with
// n = ntile*16 + l%16, kbase = ktile*32 + (l/16)*16 (contiguous k order, same as the A loads).
// kn=0: src is [N][K] row-major (weight [out][in]); kn=1: src is [K][N].
// ---------------------------------------------------------------------------
__global__ void repack_b(const float* __restrict__ src, int N, int K, int kn,
                         unsigned short* __restrict__ dst) {
    int KT = K >> 5;
    int total = (N >> 4) * KT * 32;
    int tid = blockIdx.x * blockDim.x + threadIdx.x;
    if (tid >= total) return;
    int lane  = tid & 31;
    int f     = tid >> 5;
    int ntile = f / KT;
    int ktile = f - ntile * KT;
    int n     = ntile * 16 + (lane & 15);
    int kbase = ktile * 32 + (lane >> 4) * 16;
    unsigned short* out = dst + (size_t)tid * 16;
    if (kn) {
        for (int j = 0; j < 16; ++j) out[j] = f2bf(src[(size_t)(kbase + j) * N + n]);
    } else {
        const float* p = src + (size_t)n * K + kbase;
        for (int j = 0; j < 16; ++j) out[j] = f2bf(p[j]);
    }
}

// ---------------------------------------------------------------------------
// Fused kernel: 32 rows per block, 512 threads (16 wave32).
// qk -> softmax attention -> retrieval -> gate/out GEMMs -> blend -> LN.
// ---------------------------------------------------------------------------
__global__ __launch_bounds__(THREADS) void fused_episodic(
    const float* __restrict__ x,
    const float* __restrict__ gate_b,
    const float* __restrict__ out_b,
    const float* __restrict__ ln_g,
    const float* __restrict__ ln_b,
    const float* __restrict__ keys_pos,
    const float* __restrict__ biasv,
    const unsigned short* __restrict__ keyw_rep,
    const unsigned short* __restrict__ memv_rep,
    const unsigned short* __restrict__ gate_rep,
    const unsigned short* __restrict__ out_rep,
    float* __restrict__ outp) {
    __shared__ __align__(16) unsigned short comb[MTILE * COMBS]; // [x|retr] bf16, padded
    __shared__ __align__(16) float          rt[MTILE * HID];     // retr f32 -> y in place
    __shared__ float                        qks[MTILE * KD];
    __shared__ float                        sal[MTILE * SLOTS];
    __shared__ __align__(16) unsigned short attn[MTILE * ATTNS];

    const int tid  = threadIdx.x;
    const int lane = tid & 31;
    const int w    = tid >> 5;
    const int m16  = lane & 15;
    const int hh   = lane >> 4;
    const size_t row0 = (size_t)blockIdx.x * MTILE;

    // ---- 1) x tile -> comb[:, 0:HID] as bf16 (packed b64 stores) ----------
    {
        const float4* xs = (const float4*)(x + row0 * HID);
        for (int i = tid; i < MTILE * HID / 4; i += THREADS) {
            float4 v = xs[i];
            int base = i * 4;
            int r = base >> 10;          // / HID
            int c = base & (HID - 1);
            uint2 p;
            p.x = pack2bf(v.x, v.y);
            p.y = pack2bf(v.z, v.w);
            *(uint2*)&comb[r * COMBS + c] = p;
        }
    }
    __syncthreads();

    // ---- 2) qk = xtile @ key_w^T  [32,32]  (waves 0..3, double-buffered) --
    if (w < 4) {
        v8f acc = {};
        const int mt    = w >> 1;
        const int ntile = w & 1;
        const int arow  = (mt * 16 + m16) * COMBS + hh * 16;
        const unsigned short* kwb = keyw_rep + (size_t)ntile * (HID / 32) * 32 * 16;

        BFrag a[2], b[2];
        a[0].q[0] = *(const uint4*)&comb[arow];
        a[0].q[1] = *(const uint4*)&comb[arow + 8];
        {
            const uint4* bp = (const uint4*)(kwb + (size_t)lane * 16);
            b[0].q[0] = bp[0]; b[0].q[1] = bp[1];
        }
        #pragma unroll 2
        for (int kt = 0; kt < HID / 32 - 1; ++kt) {
            const int cur = kt & 1, nxt = cur ^ 1;
            int ka = (kt + 1) * 32;
            a[nxt].q[0] = *(const uint4*)&comb[arow + ka];
            a[nxt].q[1] = *(const uint4*)&comb[arow + ka + 8];
            const uint4* bp = (const uint4*)(kwb + ((size_t)(kt + 1) * 32 + lane) * 16);
            b[nxt].q[0] = bp[0]; b[nxt].q[1] = bp[1];
            acc = __builtin_amdgcn_wmma_f32_16x16x32_bf16(
                false, a[cur].v, false, b[cur].v, (short)0, acc, false, false);
        }
        {
            const int cur = (HID / 32 - 1) & 1;
            acc = __builtin_amdgcn_wmma_f32_16x16x32_bf16(
                false, a[cur].v, false, b[cur].v, (short)0, acc, false, false);
        }
        for (int r = 0; r < 8; ++r)
            qks[(mt * 16 + r + hh * 8) * KD + ntile * 16 + m16] = acc[r];
    }
    __syncthreads();

    // ---- 3a) salience dot-products (8 waves: row = tid&31, 8 slots each) --
    if (tid < 256) {
        const int row = tid & 31;
        const int sg  = tid >> 5;          // slot group 0..7
        float q[KD]; float ss = 0.f;
        for (int d = 0; d < KD; ++d) { float t = qks[row * KD + d]; q[d] = t; ss += t * t; }
        float inv = 1.0f / fmaxf(sqrtf(ss), 1e-12f);
        const float simscale = 0.45f * inv * 0.17677669529663688110f; // 0.45/sqrt(32)/|q|
        for (int s = sg * 8; s < sg * 8 + 8; ++s) {
            float d = 0.f;
            const float* kp = keys_pos + s * KD;
            for (int t = 0; t < KD; ++t) d += q[t] * kp[t];
            sal[row * SLOTS + s] = fminf(fmaxf(d * simscale + biasv[s], 0.f), 1.f);
        }
    }
    __syncthreads();

    // ---- 3b) per-row softmax -> attn bf16 (wave 0) ------------------------
    if (tid < MTILE) {
        float mx = -1e30f;
        for (int s = 0; s < SLOTS; ++s) mx = fmaxf(mx, sal[tid * SLOTS + s]);
        float sum = 0.f;
        float e[SLOTS];
        for (int s = 0; s < SLOTS; ++s) {
            float t = expf(sal[tid * SLOTS + s] - mx);
            e[s] = t; sum += t;
        }
        float isum = 1.0f / sum;
        for (int s = 0; s < SLOTS; ++s)
            attn[tid * ATTNS + s] = f2bf(e[s] * isum);
    }
    __syncthreads();

    // ---- 4) retrieved = attn @ mem_vals  [32,1024] ------------------------
    for (int j = 0; j < 4; ++j) {
        int n = w * 4 + j;
        v8f acc0 = {}, acc1 = {};
        for (int kt = 0; kt < SLOTS / 32; ++kt) {
            BFrag a0, a1, b;
            int ab0 = (m16)      * ATTNS + kt * 32 + hh * 16;
            int ab1 = (16 + m16) * ATTNS + kt * 32 + hh * 16;
            a0.q[0] = *(const uint4*)&attn[ab0];
            a0.q[1] = *(const uint4*)&attn[ab0 + 8];
            a1.q[0] = *(const uint4*)&attn[ab1];
            a1.q[1] = *(const uint4*)&attn[ab1 + 8];
            const uint4* bp = (const uint4*)(memv_rep +
                ((size_t)(n * (SLOTS / 32) + kt) * 32 + lane) * 16);
            b.q[0] = bp[0]; b.q[1] = bp[1];
            acc0 = __builtin_amdgcn_wmma_f32_16x16x32_bf16(
                false, a0.v, false, b.v, (short)0, acc0, false, false);
            acc1 = __builtin_amdgcn_wmma_f32_16x16x32_bf16(
                false, a1.v, false, b.v, (short)0, acc1, false, false);
        }
        int cc = n * 16 + m16;
        for (int r = 0; r < 8; ++r) {
            int rr0 = r + hh * 8, rr1 = 16 + r + hh * 8;
            float v0 = acc0[r], v1 = acc1[r];
            rt[rr0 * HID + cc] = v0;
            rt[rr1 * HID + cc] = v1;
            comb[rr0 * COMBS + HID + cc] = f2bf(v0);
            comb[rr1 * COMBS + HID + cc] = f2bf(v1);
        }
    }
    __syncthreads();

    // ---- 5) gate/out GEMMs, double-buffered K-loop, fused epilogue --------
    for (int j = 0; j < 4; ++j) {
        const int n = w * 4 + j;
        v8f accg0 = {}, accg1 = {}, acco0 = {}, acco1 = {};
        const unsigned short* gbase = gate_rep + (size_t)n * (TWO_HID / 32) * 32 * 16;
        const unsigned short* obase = out_rep  + (size_t)n * (TWO_HID / 32) * 32 * 16;
        const int ar0 = m16 * COMBS + hh * 16;
        const int ar1 = (16 + m16) * COMBS + hh * 16;

        BFrag a0[2], a1[2], bg[2], bo[2];
        // preload kt = 0
        {
            a0[0].q[0] = *(const uint4*)&comb[ar0];
            a0[0].q[1] = *(const uint4*)&comb[ar0 + 8];
            a1[0].q[0] = *(const uint4*)&comb[ar1];
            a1[0].q[1] = *(const uint4*)&comb[ar1 + 8];
            const uint4* gp = (const uint4*)(gbase + (size_t)lane * 16);
            const uint4* op = (const uint4*)(obase + (size_t)lane * 16);
            bg[0].q[0] = gp[0]; bg[0].q[1] = gp[1];
            bo[0].q[0] = op[0]; bo[0].q[1] = op[1];
        }
        #pragma unroll 2
        for (int kt = 0; kt < TWO_HID / 32 - 1; ++kt) {
            const int cur = kt & 1, nxt = cur ^ 1;
            // issue loads for kt+1 (in flight during this step's WMMAs)
            {
                int ka = (kt + 1) * 32;
                a0[nxt].q[0] = *(const uint4*)&comb[ar0 + ka];
                a0[nxt].q[1] = *(const uint4*)&comb[ar0 + ka + 8];
                a1[nxt].q[0] = *(const uint4*)&comb[ar1 + ka];
                a1[nxt].q[1] = *(const uint4*)&comb[ar1 + ka + 8];
                const uint4* gp = (const uint4*)(gbase + ((size_t)(kt + 1) * 32 + lane) * 16);
                const uint4* op = (const uint4*)(obase + ((size_t)(kt + 1) * 32 + lane) * 16);
                bg[nxt].q[0] = gp[0]; bg[nxt].q[1] = gp[1];
                bo[nxt].q[0] = op[0]; bo[nxt].q[1] = op[1];
            }
            accg0 = __builtin_amdgcn_wmma_f32_16x16x32_bf16(
                false, a0[cur].v, false, bg[cur].v, (short)0, accg0, false, false);
            acco0 = __builtin_amdgcn_wmma_f32_16x16x32_bf16(
                false, a0[cur].v, false, bo[cur].v, (short)0, acco0, false, false);
            accg1 = __builtin_amdgcn_wmma_f32_16x16x32_bf16(
                false, a1[cur].v, false, bg[cur].v, (short)0, accg1, false, false);
            acco1 = __builtin_amdgcn_wmma_f32_16x16x32_bf16(
                false, a1[cur].v, false, bo[cur].v, (short)0, acco1, false, false);
        }
        {
            const int cur = (TWO_HID / 32 - 1) & 1;
            accg0 = __builtin_amdgcn_wmma_f32_16x16x32_bf16(
                false, a0[cur].v, false, bg[cur].v, (short)0, accg0, false, false);
            acco0 = __builtin_amdgcn_wmma_f32_16x16x32_bf16(
                false, a0[cur].v, false, bo[cur].v, (short)0, acco0, false, false);
            accg1 = __builtin_amdgcn_wmma_f32_16x16x32_bf16(
                false, a1[cur].v, false, bg[cur].v, (short)0, accg1, false, false);
            acco1 = __builtin_amdgcn_wmma_f32_16x16x32_bf16(
                false, a1[cur].v, false, bo[cur].v, (short)0, acco1, false, false);
        }

        // fused epilogue: sigmoid gate, exact GELU, blend, residual
        const int cc = n * 16 + m16;
        const float gb = gate_b[cc], ob = out_b[cc];
        for (int r = 0; r < 8; ++r) {
            int rr0 = r + hh * 8;
            int rr1 = 16 + r + hh * 8;
            float g0 = 1.0f / (1.0f + expf(-(accg0[r] + gb)));
            float g1 = 1.0f / (1.0f + expf(-(accg1[r] + gb)));
            float h0 = gelu_exact(acco0[r] + ob);
            float h1 = gelu_exact(acco1[r] + ob);
            float r0 = rt[rr0 * HID + cc];
            float r1 = rt[rr1 * HID + cc];
            float x0 = x[(row0 + rr0) * HID + cc];
            float x1 = x[(row0 + rr1) * HID + cc];
            rt[rr0 * HID + cc] = h0 + g0 * r0 + (1.0f - g0) * x0;  // y in-place
            rt[rr1 * HID + cc] = h1 + g1 * r1 + (1.0f - g1) * x1;
        }
    }
    __syncthreads();

    // ---- 6) LayerNorm, 2 rows per wave ------------------------------------
    for (int rr = w * 2; rr < w * 2 + 2; ++rr) {
        float s = 0.f, s2 = 0.f;
        for (int i = lane; i < HID; i += 32) {
            float v = rt[rr * HID + i];
            s += v; s2 += v * v;
        }
        for (int m = 16; m >= 1; m >>= 1) {
            s  += __shfl_xor(s,  m, 32);
            s2 += __shfl_xor(s2, m, 32);
        }
        float mu  = s  * (1.0f / HID);
        float var = s2 * (1.0f / HID) - mu * mu;
        float inv = rsqrtf(var + 1e-5f);
        for (int i = lane; i < HID; i += 32) {
            float v = (rt[rr * HID + i] - mu) * inv;
            outp[(row0 + rr) * HID + i] = v * ln_g[i] + ln_b[i];
        }
    }
}

// ---------------------------------------------------------------------------
extern "C" void kernel_launch(void* const* d_in, const int* in_sizes, int n_in,
                              void* d_out, int out_size, void* d_ws, size_t ws_size,
                              hipStream_t stream) {
    const float* x         = (const float*)d_in[0];
    const float* key_w     = (const float*)d_in[1];
    const float* out_w     = (const float*)d_in[2];
    const float* out_b     = (const float*)d_in[3];
    const float* gate_w    = (const float*)d_in[4];
    const float* gate_b    = (const float*)d_in[5];
    const float* ln_g      = (const float*)d_in[6];
    const float* ln_b      = (const float*)d_in[7];
    const float* pos_table = (const float*)d_in[8];
    const float* mem_keys  = (const float*)d_in[9];
    const float* mem_vals  = (const float*)d_in[10];
    const float* mem_age   = (const float*)d_in[11];
    const float* mem_conf  = (const float*)d_in[12];
    const int*   slot_order= (const int*)d_in[13];
    float* outp = (float*)d_out;

    const int nrows = in_sizes[0] / HID;   // B

    // workspace carve-out (~8.6 MB)
    char* ws = (char*)d_ws;
    size_t off = 0;
    auto carve = [&](size_t bytes) -> char* {
        char* p = ws + off;
        off = (off + bytes + 255) & ~(size_t)255;
        return p;
    };
    float* keys_pos          = (float*)carve((size_t)SLOTS * KD * 4);
    float* biasv             = (float*)carve((size_t)SLOTS * 4);
    unsigned short* keyw_rep = (unsigned short*)carve((size_t)KD * HID * 2);
    unsigned short* memv_rep = (unsigned short*)carve((size_t)SLOTS * HID * 2);
    unsigned short* gate_rep = (unsigned short*)carve((size_t)HID * TWO_HID * 2);
    unsigned short* out_rep  = (unsigned short*)carve((size_t)HID * TWO_HID * 2);

    prep_keys_bias<<<1, SLOTS, 0, stream>>>(pos_table, mem_keys, mem_age,
                                            mem_conf, slot_order, keys_pos, biasv);

    int t;
    t = (KD / 16) * (HID / 32) * 32;          // key_w  [N=32][K=1024]
    repack_b<<<(t + 255) / 256, 256, 0, stream>>>(key_w, KD, HID, 0, keyw_rep);
    t = (HID / 16) * (SLOTS / 32) * 32;       // mem_vals [K=64][N=1024]
    repack_b<<<(t + 255) / 256, 256, 0, stream>>>(mem_vals, HID, SLOTS, 1, memv_rep);
    t = (HID / 16) * (TWO_HID / 32) * 32;     // gate_w / out_w [N=1024][K=2048]
    repack_b<<<(t + 255) / 256, 256, 0, stream>>>(gate_w, HID, TWO_HID, 0, gate_rep);
    repack_b<<<(t + 255) / 256, 256, 0, stream>>>(out_w, HID, TWO_HID, 0, out_rep);

    fused_episodic<<<nrows / MTILE, THREADS, 0, stream>>>(
        x, gate_b, out_b, ln_g, ln_b, keys_pos, biasv,
        keyw_rep, memv_rep, gate_rep, out_rep, outp);
}
